// MAGECActor_26852135535308
// MI455X (gfx1250) — compile-verified, matrix operand order checked by reading, not run on hardware
//
#include <hip/hip_runtime.h>
#include <hip/hip_bf16.h>

typedef __attribute__((ext_vector_type(16))) _Float16 v16h;
typedef __attribute__((ext_vector_type(8)))  float    v8f;
typedef __attribute__((ext_vector_type(8)))  _Float16 h8;
typedef int i4v __attribute__((vector_size(4 * sizeof(int))));

constexpr int B_  = 8;
constexpr int N_  = 10000;
constexpr int E_  = 320000;
constexpr int H_  = 128;

// ---------------------------------------------------------------------------
// gfx1250 async global->LDS path (ASYNCcnt-tracked). The builtin takes
// (int4 AS(1)* src, int4 AS(3)* dst, imm offset, imm cpol) per the compiler's
// own diagnostic. Guarded so code still compiles if absent.
// ---------------------------------------------------------------------------
#if defined(__has_builtin)
#if __has_builtin(__builtin_amdgcn_global_load_async_to_lds_b128) && \
    __has_builtin(__builtin_amdgcn_s_wait_asynccnt)
#define USE_ASYNC_LDS 1
#endif
#endif

__device__ inline void g2l_16B(void* ldsDst, const void* gSrc) {
#ifdef USE_ASYNC_LDS
  __builtin_amdgcn_global_load_async_to_lds_b128(
      (__attribute__((address_space(1))) i4v*)gSrc,
      (__attribute__((address_space(3))) i4v*)ldsDst, 0, 0);
#else
  *(h8*)ldsDst = *(const h8*)gSrc;
#endif
}
__device__ inline void g2l_wait() {
#ifdef USE_ASYNC_LDS
  __builtin_amdgcn_s_wait_asynccnt(0);
#endif
}

// ---------------------------------------------------------------------------
// WMMA helpers (gfx1250 wave32, 16x16x32 f16 -> f32)
// ---------------------------------------------------------------------------
__device__ inline v8f wmma_f16(v16h a, v16h b, v8f c) {
  return __builtin_amdgcn_wmma_f32_16x16x32_f16(
      /*neg_a=*/false, a, /*neg_b=*/false, b,
      /*c_mod=*/(short)0, c, /*reuse_a=*/false, /*reuse_b=*/false);
}

// Load one 16x32 (f16) operand fragment from an LDS tile laid out [16][ld].
// ISA layout: lane m=lane&15 holds row m; k-half kh=lane>>4 selects
// K {0..7,16..23} (kh=0) or {8..15,24..31} (kh=1) -> two b128 LDS loads.
__device__ inline v16h lds_frag(const _Float16* base, int ld, int lane) {
  const int m  = lane & 15;
  const int kh = (lane >> 4) & 1;
  const h8* p0 = (const h8*)(base + m * ld + 8 * kh);
  const h8* p1 = (const h8*)(base + m * ld + 16 + 8 * kh);
  h8 a0 = *p0;
  h8 a1 = *p1;
  v16h r;
#pragma unroll
  for (int i = 0; i < 8; ++i) { r[i] = a0[i]; r[i + 8] = a1[i]; }
  return r;
}

// ---------------------------------------------------------------------------
// Weight packing: fp32 row-major [K][N] -> fp16 transposed [N][Kpad]
// ---------------------------------------------------------------------------
__global__ void pack_weights(const float* __restrict__ W_in,
                             const float* __restrict__ W_nb,
                             const float* __restrict__ W_self,
                             _Float16* __restrict__ WinT,
                             _Float16* __restrict__ WnbT,
                             _Float16* __restrict__ WselfT) {
  const int T0 = 128 * 64;        // W_in^T
  const int T1 = 3 * 128 * 160;   // W_nb^T  (K padded 144->160)
  const int T2 = 3 * 128 * 128;   // W_self^T
  for (int idx = blockIdx.x * blockDim.x + threadIdx.x; idx < T0 + T1 + T2;
       idx += gridDim.x * blockDim.x) {
    if (idx < T0) {
      int n = idx / 64, k = idx % 64;
      WinT[idx] = (_Float16)W_in[k * 128 + n];
    } else if (idx < T0 + T1) {
      int j = idx - T0;
      int l = j / (128 * 160), r = j % (128 * 160);
      int n = r / 160, k = r % 160;
      float v = (k < 144) ? W_nb[((size_t)l * 144 + k) * 128 + n] : 0.0f;
      WnbT[j] = (_Float16)v;
    } else {
      int j = idx - T0 - T1;
      int l = j / (128 * 128), r = j % (128 * 128);
      int n = r / 128, k = r % 128;
      WselfT[j] = (_Float16)W_self[((size_t)l * 128 + k) * 128 + n];
    }
  }
}

__global__ void fill_f32(float* __restrict__ p, size_t n, float v) {
  size_t stride = (size_t)gridDim.x * blockDim.x;
  for (size_t i = (size_t)blockIdx.x * blockDim.x + threadIdx.x; i < n; i += stride)
    p[i] = v;
}

// Degree (destination counts). cnt is pre-filled with 1.0 (self-loop).
__global__ void count_deg(const int* __restrict__ eidx, float* __restrict__ cnt) {
  size_t tot = (size_t)B_ * E_;
  size_t stride = (size_t)gridDim.x * blockDim.x;
  for (size_t i = (size_t)blockIdx.x * blockDim.x + threadIdx.x; i < tot; i += stride) {
    int b = (int)(i / E_);
    int e = (int)(i % E_);
    int d = eidx[(size_t)b * 2 * E_ + E_ + e];
    atomicAdd(&cnt[(size_t)b * N_ + d], 1.0f);
  }
}

// ---------------------------------------------------------------------------
// Input projection: h0 = x @ W_in + b_in   (N x 64) @ (64 x 128) -> f16
// Output bounced through LDS so global stores are coalesced b128.
// ---------------------------------------------------------------------------
__global__ __launch_bounds__(256)
void proj_kernel(const float* __restrict__ x, const float* __restrict__ b_in,
                 const _Float16* __restrict__ WinT, _Float16* __restrict__ h0) {
  __shared__ __align__(16) _Float16 sW[128 * 64];   // 16 KB
  __shared__ __align__(16) _Float16 sX[16 * 64];    // 2 KB
  __shared__ __align__(16) _Float16 sO[16 * 128];   // 4 KB
  const int t = threadIdx.x;
  for (int i = t; i < 128 * 64 / 8; i += 256) ((h8*)sW)[i] = ((const h8*)WinT)[i];
  __syncthreads();
  const int wave = t >> 5, lane = t & 31;
  const int nTiles = (N_ + 15) / 16;
  for (int tile = blockIdx.x; tile < B_ * nTiles; tile += gridDim.x) {
    int b = tile / nTiles, rt = tile % nTiles;
    for (int i = t; i < 16 * 64; i += 256) {
      int r = i >> 6, k = i & 63;
      int g = rt * 16 + r;
      sX[i] = (_Float16)((g < N_) ? x[((size_t)b * N_ + g) * 64 + k] : 0.0f);
    }
    __syncthreads();
    float bias = b_in[wave * 16 + (lane & 15)];
    v8f c;
#pragma unroll
    for (int i = 0; i < 8; ++i) c[i] = bias;
#pragma unroll
    for (int ks = 0; ks < 2; ++ks) {
      v16h a  = lds_frag(sX + ks * 32, 64, lane);
      v16h bb = lds_frag(sW + wave * 16 * 64 + ks * 32, 64, lane);
      c = wmma_f16(a, bb, c);
    }
#pragma unroll
    for (int i = 0; i < 8; ++i) {
      int m = i + ((lane >> 4) << 3);
      sO[m * 128 + wave * 16 + (lane & 15)] = (_Float16)c[i];
    }
    __syncthreads();
    {  // coalesced store: one 16B chunk per thread
      int r = t / 16, cp = t % 16;
      int g = rt * 16 + r;
      if (g < N_)
        ((h8*)(h0 + ((size_t)b * N_ + g) * H_))[cp] = ((const h8*)sO)[t];
    }
    __syncthreads();
  }
}

// ---------------------------------------------------------------------------
// Edge messages (dominant kernel): fused gather -> WMMA -> atomic scatter.
// Segments: e<E real edges, else self-loops (src=dst=e-E, edge_attr=0).
// msg = [h[src] | ea] @ W_nb (bias folded into combine). K=144 padded to 160.
// ---------------------------------------------------------------------------
__global__ __launch_bounds__(256)
void edge_kernel(const int* __restrict__ eidx, const float* __restrict__ eattr,
                 const _Float16* __restrict__ hin,
                 const _Float16* __restrict__ WnbT_all, int layer,
                 float* __restrict__ agg) {
  __shared__ __align__(16) _Float16 sW[128 * 160];  // 40 KB: W_nb[l]^T
  __shared__ __align__(16) _Float16 sA[64 * 160];   // 20 KB: A tile (64 edges)
  __shared__ int sDst[64];
  const int t = threadIdx.x;
  const _Float16* Wl = WnbT_all + (size_t)layer * 128 * 160;
  for (int i = t; i < 128 * 160 / 8; i += 256)
    g2l_16B((h8*)sW + i, (const h8*)Wl + i);
  g2l_wait();
  __syncthreads();
  const int wave = t >> 5, lane = t & 31;
  const int TOT = E_ + N_;
  const int nTiles = (TOT + 63) / 64;
  for (int tile = blockIdx.x; tile < B_ * nTiles; tile += gridDim.x) {
    int b = tile / nTiles, et = tile % nTiles;
    int base = et * 64;
    // prefetch next tile's indices while we work on this one
    {
      int nt = tile + gridDim.x;
      if (nt < B_ * nTiles && t == 0) {
        int nb = nt / nTiles, ne = (nt % nTiles) * 64;
        if (ne < E_) {
          __builtin_prefetch(eidx + (size_t)nb * 2 * E_ + ne, 0, 1);
          __builtin_prefetch(eidx + (size_t)nb * 2 * E_ + E_ + ne, 0, 1);
        }
      }
    }
    {  // gather 64 h[src] rows, 4 threads per row (32 f16 each)
      int r = t >> 2, q = t & 3;
      int e = base + r;
      int src = 0;
      bool valid = false;
      if (e < E_)       { src = eidx[(size_t)b * 2 * E_ + e]; valid = true; }
      else if (e < TOT) { src = e - E_; valid = true; }
      h8* d = (h8*)(sA + r * 160 + q * 32);
      if (valid) {
        const h8* s = (const h8*)(hin + ((size_t)b * N_ + src) * H_) + q * 4;
#pragma unroll
        for (int j = 0; j < 4; ++j) g2l_16B(d + j, s + j);
      } else {
        h8 z = {};
#pragma unroll
        for (int j = 0; j < 4; ++j) d[j] = z;
      }
    }
    if (t < 64) {  // edge_attr cols 128..143 (+pad) and dst id
      int r = t, e = base + r;
      int dv = -1;
      if (e < E_) {
        dv = eidx[(size_t)b * 2 * E_ + E_ + e];
#pragma unroll
        for (int k = 0; k < 16; ++k)
          sA[r * 160 + 128 + k] = (_Float16)eattr[((size_t)b * E_ + e) * 16 + k];
      } else {
        if (e < TOT) dv = e - E_;
#pragma unroll
        for (int k = 0; k < 16; ++k) sA[r * 160 + 128 + k] = (_Float16)0.0f;
      }
#pragma unroll
      for (int k = 16; k < 32; ++k) sA[r * 160 + 128 + k] = (_Float16)0.0f;
      sDst[r] = dv;
    }
    g2l_wait();
    __syncthreads();
    float* aggB = agg + (size_t)b * N_ * H_;
#pragma unroll
    for (int rs = 0; rs < 4; ++rs) {  // 4 row-subtiles x (this wave's col tile)
      v8f c = {};
#pragma unroll
      for (int ks = 0; ks < 5; ++ks) {  // K = 160 = 5 x 32
        v16h a  = lds_frag(sA + rs * 16 * 160 + ks * 32, 160, lane);
        v16h bb = lds_frag(sW + wave * 16 * 160 + ks * 32, 160, lane);
        c = wmma_f16(a, bb, c);
      }
#pragma unroll
      for (int i = 0; i < 8; ++i) {
        int m = i + ((lane >> 4) << 3);
        int g = sDst[rs * 16 + m];
        if (g >= 0)
          atomicAdd(&aggB[(size_t)g * H_ + wave * 16 + (lane & 15)], c[i]);
      }
    }
    __syncthreads();
  }
}

// ---------------------------------------------------------------------------
// Combine: out = agg/cnt + b_nb + h @ W_self + b_self ; L2-normalize ; ReLU
// ---------------------------------------------------------------------------
__global__ __launch_bounds__(256)
void combine_kernel(const _Float16* __restrict__ hin, const float* __restrict__ agg,
                    const float* __restrict__ cnt,
                    const _Float16* __restrict__ WselfT_all,
                    const float* __restrict__ b_self, const float* __restrict__ b_nb,
                    int layer, _Float16* __restrict__ hout) {
  __shared__ __align__(16) _Float16 sW[128 * 128];  // 32 KB
  __shared__ __align__(16) _Float16 sH[16 * 128];   // 4 KB
  __shared__ float sOut[16 * 128];                  // 8 KB
  __shared__ float sNorm[16];
  const int t = threadIdx.x;
  const _Float16* Wl = WselfT_all + (size_t)layer * 128 * 128;
  for (int i = t; i < 128 * 128 / 8; i += 256) ((h8*)sW)[i] = ((const h8*)Wl)[i];
  __syncthreads();
  const int wave = t >> 5, lane = t & 31;
  const int nTiles = (N_ + 15) / 16;
  for (int tile = blockIdx.x; tile < B_ * nTiles; tile += gridDim.x) {
    int b = tile / nTiles, rt = tile % nTiles;
    if (t < 16) sNorm[t] = 0.0f;
    for (int i = t; i < 16 * 128 / 8; i += 256) {  // stage h tile (16 h8 / row)
      int r = i / 16, cpart = i % 16;
      int g = rt * 16 + r;
      h8* d = (h8*)sH + i;
      if (g < N_) *d = ((const h8*)(hin + ((size_t)b * N_ + g) * H_))[cpart];
      else        { h8 z = {}; *d = z; }
    }
    __syncthreads();
    v8f c = {};
#pragma unroll
    for (int ks = 0; ks < 4; ++ks) {  // K = 128
      v16h a  = lds_frag(sH + ks * 32, 128, lane);
      v16h bb = lds_frag(sW + wave * 16 * 128 + ks * 32, 128, lane);
      c = wmma_f16(a, bb, c);
    }
#pragma unroll
    for (int i = 0; i < 8; ++i) {
      int m = i + ((lane >> 4) << 3);
      int g = rt * 16 + m;
      int col = wave * 16 + (lane & 15);
      float v = c[i] + b_self[layer * H_ + col] + b_nb[layer * H_ + col];
      if (g < N_) {
        float cn = cnt[(size_t)b * N_ + g];
        v += agg[((size_t)b * N_ + g) * H_ + col] / fmaxf(cn, 1.0f);
      }
      sOut[m * 128 + col] = v;
    }
    __syncthreads();
    {  // row L2 norms: 16 threads per row, 8 cols each
      int r = t >> 4, seg = t & 15;
      float p = 0.0f;
#pragma unroll
      for (int j = 0; j < 8; ++j) {
        float v = sOut[r * 128 + seg * 8 + j];
        p += v * v;
      }
      atomicAdd(&sNorm[r], p);
    }
    __syncthreads();
    {  // normalize + ReLU + emit one b128 store per thread
      int r = (t * 8) / 128;
      int c0 = (t * 8) % 128;
      int g = rt * 16 + r;
      if (g < N_) {
        float nrm = fmaxf(sqrtf(sNorm[r]), 1e-12f);
        h8 vbuf;
#pragma unroll
        for (int j = 0; j < 8; ++j) {
          float v = sOut[r * 128 + c0 + j] / nrm;
          vbuf[j] = (_Float16)fmaxf(v, 0.0f);
        }
        *(h8*)(hout + ((size_t)b * N_ + g) * H_ + c0) = vbuf;
      }
    }
    __syncthreads();
  }
}

// ---------------------------------------------------------------------------
// Head: jump GEMM for only the first M=15 rows, then the tiny MLP chain.
// One block per batch; plain VALU (work is negligible).
// ---------------------------------------------------------------------------
__global__ __launch_bounds__(256)
void head_kernel(const _Float16* __restrict__ h1, const _Float16* __restrict__ h2,
                 const _Float16* __restrict__ h3, const int* __restrict__ aidx,
                 const float* __restrict__ W_jump, const float* __restrict__ b_jump,
                 const float* __restrict__ W_s1, const float* __restrict__ b_s1,
                 const float* __restrict__ W_s2, const float* __restrict__ b_s2,
                 const float* __restrict__ W_a1, const float* __restrict__ b_a1,
                 const float* __restrict__ W_a2, const float* __restrict__ b_a2,
                 float* __restrict__ out) {
  __shared__ float sTop[15 * 128];
  __shared__ float sS[15];
  __shared__ float sScore[15];
  __shared__ float sHid2[64];
  const int b = blockIdx.x;
  const int t = threadIdx.x;
  const _Float16* hs[3] = {h1, h2, h3};
  if (t < 15) sS[t] = b_s2[0];
  for (int idx = t; idx < 15 * 128; idx += 256) {
    int row = idx >> 7, col = idx & 127;
    float acc = b_jump[col];
#pragma unroll
    for (int l = 0; l < 3; ++l) {
      const _Float16* hr = hs[l] + ((size_t)b * N_ + row) * H_;
      for (int k = 0; k < 128; ++k)
        acc += (float)hr[k] * W_jump[((size_t)(l * 128 + k)) * 128 + col];
    }
    sTop[idx] = acc;
  }
  __syncthreads();
  for (int idx = t; idx < 15 * 64; idx += 256) {
    int row = idx / 64, j = idx % 64;
    float hv = b_s1[j];
    for (int k = 0; k < 128; ++k) hv += sTop[row * 128 + k] * W_s1[k * 64 + j];
    hv = fmaxf(hv, 0.0f);
    atomicAdd(&sS[row], hv * W_s2[j]);
  }
  __syncthreads();
  if (t < 15) {
    int a = aidx[b];
    sScore[t] = (t != a) ? sS[t] : -10.0f;  // t < N always (M << N)
  }
  __syncthreads();
  if (t < 64) {
    float hv = b_a1[t];
#pragma unroll
    for (int m = 0; m < 15; ++m) hv += sScore[m] * W_a1[m * 64 + t];
    sHid2[t] = fmaxf(hv, 0.0f);
  }
  __syncthreads();
  if (t < 15) {
    float o = b_a2[t];
    for (int j = 0; j < 64; ++j) o += sHid2[j] * W_a2[j * 15 + t];
    out[b * 15 + t] = o;
  }
}

// ---------------------------------------------------------------------------
extern "C" void kernel_launch(void* const* d_in, const int* in_sizes, int n_in,
                              void* d_out, int out_size, void* d_ws, size_t ws_size,
                              hipStream_t stream) {
  (void)in_sizes; (void)n_in; (void)out_size; (void)ws_size;
  const float* x      = (const float*)d_in[0];
  const int*   eidx   = (const int*)d_in[1];
  const float* eattr  = (const float*)d_in[2];
  const int*   aidx   = (const int*)d_in[3];
  const float* W_in   = (const float*)d_in[4];
  const float* b_in   = (const float*)d_in[5];
  const float* W_nb   = (const float*)d_in[6];
  const float* b_nb   = (const float*)d_in[7];
  const float* W_self = (const float*)d_in[8];
  const float* b_self = (const float*)d_in[9];
  const float* W_jump = (const float*)d_in[10];
  const float* b_jump = (const float*)d_in[11];
  const float* W_s1   = (const float*)d_in[12];
  const float* b_s1   = (const float*)d_in[13];
  const float* W_s2   = (const float*)d_in[14];
  const float* b_s2   = (const float*)d_in[15];
  const float* W_a1   = (const float*)d_in[16];
  const float* b_a1   = (const float*)d_in[17];
  const float* W_a2   = (const float*)d_in[18];
  const float* b_a2   = (const float*)d_in[19];
  float* out = (float*)d_out;

  char* ws = (char*)d_ws;
  size_t off = 0;
  auto take = [&](size_t bytes) -> void* {
    void* p = ws + off;
    off = (off + bytes + 255) & ~(size_t)255;
    return p;
  };
  _Float16* WinT   = (_Float16*)take((size_t)128 * 64 * 2);
  _Float16* WnbT   = (_Float16*)take((size_t)3 * 128 * 160 * 2);
  _Float16* WselfT = (_Float16*)take((size_t)3 * 128 * 128 * 2);
  _Float16* h0     = (_Float16*)take((size_t)B_ * N_ * H_ * 2);
  _Float16* hl[3];
  for (int i = 0; i < 3; ++i) hl[i] = (_Float16*)take((size_t)B_ * N_ * H_ * 2);
  float* agg = (float*)take((size_t)B_ * N_ * H_ * 4);
  float* cnt = (float*)take((size_t)B_ * N_ * 4);

  pack_weights<<<128, 256, 0, stream>>>(W_in, W_nb, W_self, WinT, WnbT, WselfT);
  fill_f32<<<512, 256, 0, stream>>>(cnt, (size_t)B_ * N_, 1.0f);   // self-loops
  count_deg<<<2048, 256, 0, stream>>>(eidx, cnt);
  proj_kernel<<<1024, 256, 0, stream>>>(x, b_in, WinT, h0);

  const _Float16* hin = h0;
  for (int l = 0; l < 3; ++l) {
    fill_f32<<<2048, 256, 0, stream>>>(agg, (size_t)B_ * N_ * H_, 0.0f);
    edge_kernel<<<4096, 256, 0, stream>>>(eidx, eattr, hin, WnbT, l, agg);
    combine_kernel<<<2048, 256, 0, stream>>>(hin, agg, cnt, WselfT, b_self, b_nb,
                                             l, hl[l]);
    hin = hl[l];
  }
  head_kernel<<<B_, 256, 0, stream>>>(hl[0], hl[1], hl[2], aidx, W_jump, b_jump,
                                      W_s1, b_s1, W_s2, b_s2, W_a1, b_a1, W_a2,
                                      b_a2, out);
}
